// PointerGenerator_52613349376101
// MI455X (gfx1250) — matrix operand order repaired and checked by reading, not compiled, not run on hardware
//
#include <hip/hip_runtime.h>
#include <hip/hip_bf16.h>
#include <math.h>

typedef __attribute__((ext_vector_type(16))) _Float16 v16h;
typedef __attribute__((ext_vector_type(8)))  float    v8f;

#define Mb 8
#define Tt 128
#define R  1024      // Mb*Tt rows
#define DH 512
#define DW 256
#define INSZ 768
#define G3 1536
#define ATT 256
#define DD 512       // LEN_DOC
#define VOC 50000
#define FC 1024
#define VCH 512
#define NCH 98       // ceil(50000/512)

__device__ __forceinline__ v8f wmma16(v16h a, v16h b, v8f c) {
    return __builtin_amdgcn_wmma_f32_16x16x32_f16(false, a, false, b, (short)0, c, false, false);
}

// A fragment: 16 rows x 32 K, row-major fp32 source -> f16
__device__ __forceinline__ v16h load_a(const float* __restrict__ tile, int ld, int k0, int lane) {
    const int r = lane & 15, hlf = lane >> 4;
    const float* p = tile + (size_t)r * ld + k0 + hlf * 16;
    v16h f;
#pragma unroll
    for (int i = 0; i < 8; ++i) f[i] = (_Float16)p[i];
#pragma unroll
    for (int i = 0; i < 8; ++i) f[8 + i] = (_Float16)p[8 + i];
    return f;
}
// B fragment from row-major W (N x K): tile = &W[n0*ldb]
__device__ __forceinline__ v16h load_bt(const float* __restrict__ tile, int ld, int k0, int lane) {
    const int c = lane & 15, hlf = lane >> 4;
    const float* p = tile + (size_t)c * ld + k0 + hlf * 16;
    v16h f;
#pragma unroll
    for (int i = 0; i < 16; ++i) f[i] = (_Float16)p[i];
    return f;
}
// B fragment from K-major B (K x N): tile = &B[n0]
__device__ __forceinline__ v16h load_bn(const float* __restrict__ tile, int ld, int k0, int lane) {
    const int c = lane & 15, hlf = lane >> 4;
    const float* p = tile + (size_t)(k0 + hlf * 16) * ld + c;
    v16h f;
#pragma unroll
    for (int i = 0; i < 16; ++i) f[i] = (_Float16)p[(size_t)i * ld];
    return f;
}

// ---------------- generic WMMA GEMM: C = act(A @ B(^T) + bias) -----------------
// block = 128 threads (4 waves); each wave computes a 16x64 strip (4 N-tiles).
template <bool BT, bool RELU>
__global__ void k_gemm(const float* __restrict__ A, int lda, long long sA,
                       const float* __restrict__ B, int ldb, long long sB,
                       float* __restrict__ C, int ldc, long long sC,
                       const float* __restrict__ bias, int K) {
    const int lane = threadIdx.x & 31;
    const int w    = threadIdx.x >> 5;
    const int rowbase = blockIdx.y * 16;
    const int colbase = (blockIdx.x * 4 + w) * 64;
    const float* Ab = A + (long long)blockIdx.z * sA + (size_t)rowbase * lda;
    const float* Bb = B + (long long)blockIdx.z * sB;
    float*       Cb = C + (long long)blockIdx.z * sC;

    v8f acc[4];
#pragma unroll
    for (int j = 0; j < 4; ++j)
#pragma unroll
        for (int i = 0; i < 8; ++i) acc[j][i] = 0.f;

    for (int k0 = 0; k0 < K; k0 += 32) {
        v16h a = load_a(Ab, lda, k0, lane);
#pragma unroll
        for (int j = 0; j < 4; ++j) {
            v16h b;
            if (BT) b = load_bt(Bb + (size_t)(colbase + j * 16) * ldb, ldb, k0, lane);
            else    b = load_bn(Bb + (colbase + j * 16), ldb, k0, lane);
            acc[j] = wmma16(a, b, acc[j]);
        }
    }
    const int r = lane & 15, hlf = lane >> 4;
#pragma unroll
    for (int j = 0; j < 4; ++j) {
        const int cc = colbase + j * 16 + r;
        const float bv = bias ? bias[cc] : 0.f;
#pragma unroll
        for (int i = 0; i < 8; ++i) {
            float v = acc[j][i] + bv;
            if (RELU) v = fmaxf(v, 0.f);
            Cb[(size_t)(rowbase + i + 8 * hlf) * ldc + cc] = v;
        }
    }
}

// ---------------- build dec_in = [sent_hidden | emb(dec_labels)] ----------------
__global__ void k_embed(const float* __restrict__ sh, const int* __restrict__ labels,
                        const float* __restrict__ emb, float* __restrict__ dec_in) {
    int rt = blockIdx.x, m = rt >> 7, t = rt & 127;
    int tok = labels[m * Tt + t];
    for (int c = threadIdx.x; c < INSZ; c += 256)
        dec_in[(size_t)rt * INSZ + c] =
            (c < DH) ? sh[m * DH + c] : emb[(size_t)tok * DW + (c - DH)];
}

// ---------------- GRU recurrence (one workgroup, 32 waves, WMMA) ----------------
// GI already includes b_ih. gh = h @ Whh^T computed in LDS each step.
__global__ void __launch_bounds__(1024) k_gru(const float* __restrict__ GI,
                                              const float* __restrict__ Whh,
                                              const float* __restrict__ bhh,
                                              float* __restrict__ out, int out_ld) {
    extern __shared__ float smem[];
    float* hbuf  = smem;             // 16 x 512 (rows 8..15 stay zero)
    float* ghbuf = smem + 16 * DH;   // 8 x 1536
    const int tid = threadIdx.x, lane = tid & 31, w = tid >> 5;
    for (int i = tid; i < 16 * DH; i += 1024) hbuf[i] = 0.f;
    __syncthreads();

    for (int t = 0; t < Tt; ++t) {
        v8f acc[3];
#pragma unroll
        for (int j = 0; j < 3; ++j)
#pragma unroll
            for (int i = 0; i < 8; ++i) acc[j][i] = 0.f;

        for (int k0 = 0; k0 < DH; k0 += 32) {
            v16h a = load_a(hbuf, DH, k0, lane);
#pragma unroll
            for (int j = 0; j < 3; ++j) {
                int n0 = (w * 3 + j) * 16;
                v16h b = load_bt(Whh + (size_t)n0 * DH, DH, k0, lane);
                acc[j] = wmma16(a, b, acc[j]);
            }
        }
        if ((lane >> 4) == 0) {   // lanes 0..15 hold rows 0..7
            int cl = lane & 15;
#pragma unroll
            for (int j = 0; j < 3; ++j) {
                int col = (w * 3 + j) * 16 + cl;
#pragma unroll
                for (int i = 0; i < 8; ++i) ghbuf[i * G3 + col] = acc[j][i];
            }
        }
        __syncthreads();
        // gate phase: 8 rows x 512 cols = 4096 values
        for (int v = tid; v < Mb * DH; v += 1024) {
            int row = v >> 9, col = v & (DH - 1);
            const float* gi = GI + (size_t)((row << 7) + t) * G3;
            float ghr = ghbuf[row * G3 + col]            + bhh[col];
            float ghz = ghbuf[row * G3 + col + DH]       + bhh[col + DH];
            float ghn = ghbuf[row * G3 + col + 2 * DH]   + bhh[col + 2 * DH];
            float rr = 1.f / (1.f + expf(-(gi[col] + ghr)));
            float zz = 1.f / (1.f + expf(-(gi[col + DH] + ghz)));
            float nn = tanhf(gi[col + 2 * DH] + rr * ghn);
            float hp = hbuf[row * DH + col];
            float hn = (1.f - zz) * nn + zz * hp;
            hbuf[row * DH + col] = hn;
            out[(size_t)((row << 7) + t) * out_ld + col] = hn;
        }
        __syncthreads();
    }
}

// ---------------- coverage feature add: encplus += prev_cov * W_cov ----------------
__global__ void k_covadd(float* __restrict__ ep, const float* __restrict__ prev,
                         const float* __restrict__ Wcov) {
    int idx = blockIdx.x * 256 + threadIdx.x;     // 4096*256 total
    int md = idx >> 8, a = idx & 255;
    ep[idx] += prev[md] * Wcov[a];
}

// ---------------- attention scores: sum_a tanh(encplus+dec_feat)*W_att ----------------
#define DCH 32
__global__ void k_scores(const float* __restrict__ encplus, const float* __restrict__ dfeat,
                         const float* __restrict__ Watt, float* __restrict__ S) {
    __shared__ float df[ATT], wat[ATT], chunk[DCH * ATT], red[256];
    int rt = blockIdx.x, m = rt >> 7, tid = threadIdx.x;
    df[tid]  = dfeat[(size_t)rt * ATT + tid];
    wat[tid] = Watt[tid];
    __syncthreads();
    for (int d0 = 0; d0 < DD; d0 += DCH) {
        for (int r = 0; r < DCH; ++r)
            chunk[r * ATT + tid] = encplus[(size_t)(m * DD + d0 + r) * ATT + tid];
        __syncthreads();
        int dl = tid >> 3, q = tid & 7;
        float s = 0.f;
        for (int a = q * 32; a < q * 32 + 32; ++a)
            s += tanhf(chunk[dl * ATT + a] + df[a]) * wat[a];
        red[tid] = s;
        __syncthreads();
        if (q == 0) {
            float tot = 0.f;
            for (int k = 0; k < 8; ++k) tot += red[dl * 8 + k];
            S[(size_t)rt * DD + d0 + dl] = tot;
        }
        __syncthreads();
    }
}

// ---------------- masked renormalized softmax over d (in place) ----------------
__global__ void k_softmax(float* __restrict__ S, const int* __restrict__ emask) {
    __shared__ float red[256];
    int rt = blockIdx.x, tid = threadIdx.x, m = rt >> 7;
    float* row = S + (size_t)rt * DD;
    float v0 = row[tid], v1 = row[tid + 256];
    red[tid] = fmaxf(v0, v1); __syncthreads();
    for (int o = 128; o > 0; o >>= 1) { if (tid < o) red[tid] = fmaxf(red[tid], red[tid + o]); __syncthreads(); }
    float mx = red[0]; __syncthreads();
    float e0 = expf(v0 - mx), e1 = expf(v1 - mx);
    red[tid] = e0 + e1; __syncthreads();
    for (int o = 128; o > 0; o >>= 1) { if (tid < o) red[tid] += red[tid + o]; __syncthreads(); }
    float se = red[0]; __syncthreads();
    float a0 = e0 / se * (emask[m * DD + tid] ? 1.f : 0.f);
    float a1 = e1 / se * (emask[m * DD + tid + 256] ? 1.f : 0.f);
    red[tid] = a0 + a1; __syncthreads();
    for (int o = 128; o > 0; o >>= 1) { if (tid < o) red[tid] += red[tid + o]; __syncthreads(); }
    float nrm = red[0]; if (nrm < 1e-5f) nrm = 1.f;
    row[tid] = a0 / nrm; row[tid + 256] = a1 / nrm;
}

__global__ void k_sentcov(const float* __restrict__ A, float* __restrict__ sc) {
    int idx = blockIdx.x * 256 + threadIdx.x;  // 4096
    int m = idx >> 9, d = idx & (DD - 1);
    float s = 0.f;
    for (int t = 0; t < Tt; ++t) s += A[(size_t)((m << 7) + t) * DD + d];
    sc[idx] = s;
}

__global__ void k_pgen(const float* __restrict__ cat, const float* __restrict__ Wp,
                       const float* __restrict__ bp, float* __restrict__ pg) {
    int w = threadIdx.x >> 5, lane = threadIdx.x & 31;
    int row = blockIdx.x * 8 + w;
    const float* c = cat + (size_t)row * FC;
    float s = 0.f;
    for (int k = lane; k < FC; k += 32) s += c[k] * Wp[k];
    for (int o = 16; o > 0; o >>= 1) s += __shfl_xor(s, o);
    if (lane == 0) pg[row] = 1.f / (1.f + expf(-(s + bp[0])));
}

// ---------------- vocab GEMM fused with online softmax partials ----------------
__device__ __forceinline__ void smerge(float& M, float& S, float m2, float s2) {
    if (s2 > 0.f) {
        if (m2 > M) { S = S * expf(M - m2) + s2; M = m2; }
        else         S += s2 * expf(m2 - M);
    }
}

__global__ void k_vocab(const float* __restrict__ fc1, const float* __restrict__ W2,
                        const float* __restrict__ b2, float* __restrict__ pmax,
                        float* __restrict__ psum) {
    __shared__ float smax[8][16], ssum[8][16];
    const int lane = threadIdx.x & 31, w = threadIdx.x >> 5;
    const int rt = blockIdx.y, chunk = blockIdx.x;
    const int col0 = chunk * VCH + w * 64;
    const float* Ab = fc1 + (size_t)rt * 16 * FC;
    const int r = lane & 15, hlf = lane >> 4;

    v8f acc[4];
    bool valid[4];
#pragma unroll
    for (int j = 0; j < 4; ++j) {
        valid[j] = (col0 + j * 16) < VOC;
#pragma unroll
        for (int i = 0; i < 8; ++i) acc[j][i] = 0.f;
    }
    for (int k0 = 0; k0 < FC; k0 += 32) {
        v16h a = load_a(Ab, FC, k0, lane);
#pragma unroll
        for (int j = 0; j < 4; ++j)
            if (valid[j]) {
                v16h b = load_bt(W2 + (size_t)(col0 + j * 16) * FC, FC, k0, lane);
                acc[j] = wmma16(a, b, acc[j]);
            }
    }
    float M[8], S[8];
#pragma unroll
    for (int i = 0; i < 8; ++i) {
        float mx = -1e30f;
#pragma unroll
        for (int j = 0; j < 4; ++j)
            if (valid[j]) {
                float v = acc[j][i] + b2[col0 + j * 16 + r];
                acc[j][i] = v;
                mx = fmaxf(mx, v);
            }
        mx = fmaxf(mx, __shfl_xor(mx, 1));
        mx = fmaxf(mx, __shfl_xor(mx, 2));
        mx = fmaxf(mx, __shfl_xor(mx, 4));
        mx = fmaxf(mx, __shfl_xor(mx, 8));
        float s = 0.f;
#pragma unroll
        for (int j = 0; j < 4; ++j)
            if (valid[j]) s += expf(acc[j][i] - mx);
        s += __shfl_xor(s, 1); s += __shfl_xor(s, 2);
        s += __shfl_xor(s, 4); s += __shfl_xor(s, 8);
        M[i] = mx; S[i] = s;
    }
    if (r == 0) {  // lane 0 (rows 0-7) and lane 16 (rows 8-15)
#pragma unroll
        for (int i = 0; i < 8; ++i) { smax[w][hlf * 8 + i] = M[i]; ssum[w][hlf * 8 + i] = S[i]; }
    }
    __syncthreads();
    if (threadIdx.x < 16) {
        int rr = threadIdx.x;
        float MM = -1e30f, SS = 0.f;
        for (int ww = 0; ww < 8; ++ww) smerge(MM, SS, smax[ww][rr], ssum[ww][rr]);
        int row = rt * 16 + rr;
        pmax[(size_t)row * NCH + chunk] = MM;
        psum[(size_t)row * NCH + chunk] = SS;
    }
}

__global__ void k_vred(const float* __restrict__ pmax, const float* __restrict__ psum,
                       float* __restrict__ rmax, float* __restrict__ rsum) {
    __shared__ float sm[128], ss[128];
    int row = blockIdx.x, t = threadIdx.x;
    float M = -1e30f, S = 0.f;
    for (int c = t; c < NCH; c += 128)
        smerge(M, S, pmax[(size_t)row * NCH + c], psum[(size_t)row * NCH + c]);
    sm[t] = M; ss[t] = S; __syncthreads();
    for (int o = 64; o > 0; o >>= 1) {
        if (t < o) { float M2 = sm[t], S2 = ss[t]; smerge(M2, S2, sm[t + o], ss[t + o]); sm[t] = M2; ss[t] = S2; }
        __syncthreads();
    }
    if (t == 0) { rmax[row] = sm[0]; rsum[row] = ss[0]; }
}

__global__ void k_ltgt(const float* __restrict__ fc1, const float* __restrict__ W2,
                       const float* __restrict__ b2, const int* __restrict__ labels,
                       float* __restrict__ lt) {
    int w = threadIdx.x >> 5, lane = threadIdx.x & 31;
    int row = blockIdx.x * 8 + w;
    int t = row & 127, m = row >> 7;
    int tgt = (t < Tt - 1) ? labels[m * Tt + t + 1] : 0;
    const float* a  = fc1 + (size_t)row * FC;
    const float* wv = W2 + (size_t)tgt * FC;
    float s = 0.f;
    for (int k = lane; k < FC; k += 32) s += a[k] * wv[k];
    for (int o = 16; o > 0; o >>= 1) s += __shfl_xor(s, o);
    if (lane == 0) lt[row] = s + b2[tgt];
}

__global__ void __launch_bounds__(1024) k_loss(const float* __restrict__ lt,
        const float* __restrict__ rmax, const float* __restrict__ rsum,
        const float* __restrict__ pg, const float* __restrict__ A,
        const int* __restrict__ labels, const int* __restrict__ dmask,
        const int* __restrict__ iidx, float* __restrict__ out) {
    __shared__ float sl[1024], sc[1024];
    int tid = threadIdx.x, t = tid & 127, m = tid >> 7;
    float lacc = 0.f, lcnt = 0.f;
    if (t < Tt - 1) {
        float mk = dmask[m * Tt + t] ? 1.f : 0.f;
        int tgt = labels[m * Tt + t + 1];
        float p = pg[tid];
        float pv = p * expf(lt[tid] - rmax[tid]) / rsum[tid];
        float attn = 0.f;
        const float* Ar = A + (size_t)tid * DD;
        const int* ii = iidx + m * DD;
        for (int d = 0; d < DD; ++d)
            if (ii[d] == tgt) attn += Ar[d];
        attn *= (1.f - p);
        lacc = -logf(pv + attn + 1e-9f) * mk;
        lcnt = mk;
    }
    sl[tid] = lacc; sc[tid] = lcnt; __syncthreads();
    for (int o = 512; o > 0; o >>= 1) {
        if (tid < o) { sl[tid] += sl[tid + o]; sc[tid] += sc[tid + o]; }
        __syncthreads();
    }
    if (tid == 0) out[0] = sl[0] / sc[0];
}

__global__ void __launch_bounds__(1024) k_cov(const float* __restrict__ sc,
                                              const float* __restrict__ prev,
                                              float* __restrict__ out) {
    __shared__ float red[1024];
    int tid = threadIdx.x;
    float local = 0.f;
    for (int i = 0; i < 4; ++i) {
        int idx = tid + i * 1024;
        float s = sc[idx], pv = prev[idx];
        out[2 + idx] = pv + s;
        local += fminf(s, pv);
    }
    red[tid] = local; __syncthreads();
    for (int o = 512; o > 0; o >>= 1) { if (tid < o) red[tid] += red[tid + o]; __syncthreads(); }
    if (tid == 0) out[1] = red[0] / (float)(Mb * Tt);
}

extern "C" void kernel_launch(void* const* d_in, const int* in_sizes, int n_in,
                              void* d_out, int out_size, void* d_ws, size_t ws_size,
                              hipStream_t stream) {
    const float* sent_hidden = (const float*)d_in[0];
    const int*   input_index = (const int*)d_in[1];
    const float* enc_hiddens = (const float*)d_in[2];
    const int*   enc_masks   = (const int*)d_in[3];
    const int*   dec_labels  = (const int*)d_in[4];
    const int*   dec_masks   = (const int*)d_in[5];
    const float* prev_cov    = (const float*)d_in[6];
    const float* embedding   = (const float*)d_in[7];
    const float* W_ih0 = (const float*)d_in[8];
    const float* W_hh0 = (const float*)d_in[9];
    const float* b_ih0 = (const float*)d_in[10];
    const float* b_hh0 = (const float*)d_in[11];
    const float* W_ih1 = (const float*)d_in[12];
    const float* W_hh1 = (const float*)d_in[13];
    const float* b_ih1 = (const float*)d_in[14];
    const float* b_hh1 = (const float*)d_in[15];
    const float* W_out1 = (const float*)d_in[16];
    const float* b_out1 = (const float*)d_in[17];
    const float* W_out2 = (const float*)d_in[18];
    const float* b_out2 = (const float*)d_in[19];
    const float* W_pgen = (const float*)d_in[20];
    const float* b_pgen = (const float*)d_in[21];
    const float* W_enc  = (const float*)d_in[22];
    const float* W_dec  = (const float*)d_in[23];
    const float* W_att  = (const float*)d_in[24];
    const float* W_cov  = (const float*)d_in[25];
    float* out = (float*)d_out;

    float* ws      = (float*)d_ws;
    float* dec_in  = ws;                         // 1024*768
    float* GIbuf   = dec_in + (size_t)R * INSZ;  // 1024*1536 (reused for both layers)
    float* h1      = GIbuf + (size_t)R * G3;     // 1024*512
    float* cat     = h1 + (size_t)R * DH;        // 1024*1024 [encw | dec_out]
    float* encplus = cat + (size_t)R * FC;       // 4096*256
    float* dfeat   = encplus + (size_t)Mb * DD * ATT; // 1024*256
    float* Abuf    = dfeat + (size_t)R * ATT;    // 1024*512 (scores -> A in place)
    float* scov    = Abuf + (size_t)R * DD;      // 4096
    float* pgbuf   = scov + Mb * DD;             // 1024
    float* fc1b    = pgbuf + R;                  // 1024*1024
    float* pmax    = fc1b + (size_t)R * FC;      // 1024*98
    float* psum    = pmax + (size_t)R * NCH;     // 1024*98
    float* rmax    = psum + (size_t)R * NCH;     // 1024
    float* rsum    = rmax + R;                   // 1024
    float* ltb     = rsum + R;                   // 1024

    // 1. dec_in = [sent_hidden | emb]
    k_embed<<<R, 256, 0, stream>>>(sent_hidden, dec_labels, embedding, dec_in);
    // 2. GI0 = dec_in @ W_ih0^T + b_ih0
    k_gemm<true, false><<<dim3(G3 / 256, R / 16, 1), 128, 0, stream>>>(
        dec_in, INSZ, 0, W_ih0, INSZ, 0, GIbuf, G3, 0, b_ih0, INSZ);
    // 3. GRU layer 0 -> h1
    k_gru<<<1, 1024, (16 * DH + Mb * G3) * sizeof(float), stream>>>(GIbuf, W_hh0, b_hh0, h1, DH);
    // 4. GI1 = h1 @ W_ih1^T + b_ih1
    k_gemm<true, false><<<dim3(G3 / 256, R / 16, 1), 128, 0, stream>>>(
        h1, DH, 0, W_ih1, DH, 0, GIbuf, G3, 0, b_ih1, DH);
    // 5. GRU layer 1 -> dec_out (stored in cat[:,512:])
    k_gru<<<1, 1024, (16 * DH + Mb * G3) * sizeof(float), stream>>>(GIbuf, W_hh1, b_hh1, cat + DH, FC);
    // 6. enc_feat = enc_hiddens @ W_enc^T
    k_gemm<true, false><<<dim3(1, (Mb * DD) / 16, 1), 128, 0, stream>>>(
        enc_hiddens, DH, 0, W_enc, DH, 0, encplus, ATT, 0, nullptr, DH);
    // 7. + coverage feature
    k_covadd<<<(Mb * DD * ATT) / 256, 256, 0, stream>>>(encplus, prev_cov, W_cov);
    // 8. dec_feat = dec_out @ W_dec^T
    k_gemm<true, false><<<dim3(1, R / 16, 1), 128, 0, stream>>>(
        cat + DH, FC, 0, W_dec, DH, 0, dfeat, ATT, 0, nullptr, DH);
    // 9. scores, 10. masked softmax (in place), 11. sent_cov
    k_scores<<<R, 256, 0, stream>>>(encplus, dfeat, W_att, Abuf);
    k_softmax<<<R, 256, 0, stream>>>(Abuf, enc_masks);
    k_sentcov<<<(Mb * DD) / 256, 256, 0, stream>>>(Abuf, scov);
    // 12. enc_weighted = A @ enc_hiddens (batched over m) -> cat[:,0:512]
    k_gemm<false, false><<<dim3(DH / 256, Tt / 16, Mb), 128, 0, stream>>>(
        Abuf, DD, (long long)Tt * DD, enc_hiddens, DH, (long long)DD * DH,
        cat, FC, (long long)Tt * FC, nullptr, DD);
    // 13. p_gen
    k_pgen<<<R / 8, 256, 0, stream>>>(cat, W_pgen, b_pgen, pgbuf);
    // 14. fc1 = relu(cat @ W_out1^T + b_out1)
    k_gemm<true, true><<<dim3(FC / 256, R / 16, 1), 128, 0, stream>>>(
        cat, FC, 0, W_out1, FC, 0, fc1b, FC, 0, b_out1, FC);
    // 15. fused vocab GEMM + online softmax partials; 16. merge; 17. target logits
    k_vocab<<<dim3(NCH, R / 16, 1), 256, 0, stream>>>(fc1b, W_out2, b_out2, pmax, psum);
    k_vred<<<R, 128, 0, stream>>>(pmax, psum, rmax, rsum);
    k_ltgt<<<R / 8, 256, 0, stream>>>(fc1b, W_out2, b_out2, dec_labels, ltb);
    // 18. NLL loss, 19. coverage loss + final_coverage
    k_loss<<<1, 1024, 0, stream>>>(ltb, rmax, rsum, pgbuf, Abuf, dec_labels,
                                   dec_masks, input_index, out);
    k_cov<<<1, 1024, 0, stream>>>(scov, prev_cov, out);

    (void)in_sizes; (void)n_in; (void)out_size; (void)ws_size;
}
